// GraphSage_net_11751030521987
// MI455X (gfx1250) — matrix operand order repaired and checked by reading, not compiled, not run on hardware
//
#include <hip/hip_runtime.h>
#include <hip/hip_bf16.h>

// ---------------------------------------------------------------------------
// GraphSage forward on MI455X (gfx1250, wave32, WMMA).
// Aggregation commutes with the linear layer: W @ mean(gather(F))^T =
// mean(gather_cols(W @ F^T)).  So every _layer becomes two dense f16 WMMA
// GEMMs (f32 accumulate) + an S=10 gather-sum + relu epilogue.
// ---------------------------------------------------------------------------

#define APN   105
#define UEN   4000
#define HH    128
#define SS    10
#define APNP  112      // APN padded to multiple of 16
#define K105  128      // 105-long K dims padded to multiple of 32
#define F2    256      // 2*H
#define BN_EPS 1e-5f
#define INV11 (1.0f / 11.0f)

typedef __attribute__((ext_vector_type(16))) _Float16 v16h;
typedef __attribute__((ext_vector_type(8)))  _Float16 v8h;
typedef __attribute__((ext_vector_type(8)))  float    v8f;

// ---------------------------------------------------------------------------
// Generic WMMA GEMM:  C[M, ldc] = A[M,K] * B[N,K]^T (+rowBias[m] +colBias[n])
// A, B row-major f16 with K contiguous (K % 32 == 0).  One wave per 16x16
// output tile; blockDim.x = 32*(M/16); gridDim.x = Npad/16.
// Fragment layouts follow CDNA5 ISA 7.12.2 (16-bit A 16x32, B 32x16, f32 C/D).
// ---------------------------------------------------------------------------
__global__ void k_gemm_abt_wmma(const _Float16* __restrict__ A,
                                const _Float16* __restrict__ B,
                                float* __restrict__ C, int K, int ldc,
                                const float* __restrict__ rowBias,
                                const float* __restrict__ colBias) {
  const int lane = threadIdx.x & 31;
  const int wave = threadIdx.x >> 5;
  const int m0 = wave << 4;
  const int n0 = blockIdx.x << 4;

  // A frag: lanes 0-15 hold row m0+lane, K = {0..7, 16..23}; lanes 16-31 the
  //         same rows at K = {8..15, 24..31}.
  const _Float16* ap = A + (size_t)(m0 + (lane & 15)) * K + ((lane >> 4) << 3);
  // B frag: lanes 0-15 hold col n0+lane, K = {0..7, 16..23}; lanes 16-31 the
  //         same cols at K = {8..15, 24..31}.
  const _Float16* bp = B + (size_t)(n0 + (lane & 15)) * K + ((lane >> 4) << 3);

  v8f acc = {};
  for (int k = 0; k < K; k += 32) {
    v8h a0 = *(const v8h*)(ap + k);
    v8h a1 = *(const v8h*)(ap + k + 16);
    v8h b0 = *(const v8h*)(bp + k);
    v8h b1 = *(const v8h*)(bp + k + 16);
    v16h a = __builtin_shufflevector(a0, a1, 0, 1, 2, 3, 4, 5, 6, 7,
                                     8, 9, 10, 11, 12, 13, 14, 15);
    v16h b = __builtin_shufflevector(b0, b1, 0, 1, 2, 3, 4, 5, 6, 7,
                                     8, 9, 10, 11, 12, 13, 14, 15);
    acc = __builtin_amdgcn_wmma_f32_16x16x32_f16(
        /*neg_a=*/false, a, /*neg_b=*/false, b,
        /*c_mod=*/(short)0, acc, /*reuse_a=*/false, /*reuse_b=*/false);
  }

  // D layout: VGPR j holds M = m0 + (lane>=16 ? 8 : 0) + j, N = n0 + lane%16.
  const int n = n0 + (lane & 15);
  const int mb = m0 + ((lane >> 4) << 3);
  float cb = colBias ? colBias[n] : 0.0f;
#pragma unroll
  for (int j = 0; j < 8; ++j) {
    float v = acc[j] + cb;
    if (rowBias) v += rowBias[mb + j];
    C[(size_t)(mb + j) * ldc + n] = v;
  }
}

// ---------------------------------------------------------------------------
// two-hop index: out[n,s] = sec[ one[n,s]*S + s ]
// ---------------------------------------------------------------------------
__global__ void k_two_hop(const int* __restrict__ one, const int* __restrict__ sec,
                          int n_total, int* __restrict__ out) {
  int t = blockIdx.x * blockDim.x + threadIdx.x;
  if (t >= n_total) return;
  int s = t % SS;
  out[t] = sec[(size_t)one[t] * SS + s];
}

// dst[r, c] (dstRows x dstCols, f16, zero padded) = src[r, col0 + c]
__global__ void k_stage_f16(const float* __restrict__ src, int srcLd, int col0,
                            _Float16* __restrict__ dst, int rows, int cols,
                            int dstRows, int dstCols) {
  int t = blockIdx.x * blockDim.x + threadIdx.x;
  if (t >= dstRows * dstCols) return;
  int r = t / dstCols, c = t % dstCols;
  float v = (r < rows && c < cols) ? src[(size_t)r * srcLd + col0 + c] : 0.0f;
  dst[t] = (_Float16)v;
}

// dst[r, c] = src[c, r]  (transpose-stage, zero padded)
__global__ void k_stageT_f16(const float* __restrict__ src, int srcRows, int srcCols,
                             _Float16* __restrict__ dst, int dstRows, int dstCols) {
  int t = blockIdx.x * blockDim.x + threadIdx.x;
  if (t >= dstRows * dstCols) return;
  int r = t / dstCols, c = t % dstCols;
  float v = (c < srcRows && r < srcCols) ? src[(size_t)c * srcCols + r] : 0.0f;
  dst[t] = (_Float16)v;
}

// dst[n, f] = f<128 ? xa[f*ldx+n] : xb[(f-128)*ldx+n]   (f16 feature matrix)
__global__ void k_stage_concatT(const float* __restrict__ xa,
                                const float* __restrict__ xb, int ldx,
                                _Float16* __restrict__ dst, int rows) {
  int t = blockIdx.x * blockDim.x + threadIdx.x;
  if (t >= rows * F2) return;
  int n = t / F2, f = t % F2;
  float v = (f < HH) ? xa[(size_t)f * ldx + n] : xb[(size_t)(f - HH) * ldx + n];
  dst[t] = (_Float16)v;
}

// out[m,n] = relu( G[m,n] + (1/11) * sum_s P[m, idx[n*S+s]] ), 0 in pad cols
__global__ void k_combine_gather(const float* __restrict__ G,
                                 const float* __restrict__ P, int ldP,
                                 const int* __restrict__ idx,
                                 float* __restrict__ out, int ldN, int Nvalid) {
  int t = blockIdx.x * blockDim.x + threadIdx.x;
  if (t >= HH * ldN) return;
  int m = t / ldN, n = t % ldN;
  float r = 0.0f;
  if (n < Nvalid) {
    float s = 0.0f;
#pragma unroll
    for (int q = 0; q < SS; ++q) s += P[(size_t)m * ldP + idx[n * SS + q]];
    r = fmaxf(G[t] + INV11 * s, 0.0f);
  }
  out[t] = r;
}

// Per-feature BN stats over u<UEN of xt[u,f] = (f<128 ? x7 : x8)[f, u]
__global__ void k_bn_stats(const float* __restrict__ x7, const float* __restrict__ x8,
                           const float* __restrict__ gamma, const float* __restrict__ beta,
                           float* __restrict__ scale, float* __restrict__ shift) {
  __shared__ float s1[256], s2[256];
  int f = blockIdx.x;
  const float* row = (f < HH) ? (x7 + (size_t)f * UEN) : (x8 + (size_t)(f - HH) * UEN);
  float a = 0.0f, b = 0.0f;
  for (int u = threadIdx.x; u < UEN; u += blockDim.x) {
    float v = row[u];
    a += v; b += v * v;
  }
  s1[threadIdx.x] = a; s2[threadIdx.x] = b;
  __syncthreads();
  for (int off = 128; off > 0; off >>= 1) {
    if (threadIdx.x < off) { s1[threadIdx.x] += s1[threadIdx.x + off];
                             s2[threadIdx.x] += s2[threadIdx.x + off]; }
    __syncthreads();
  }
  if (threadIdx.x == 0) {
    float mean = s1[0] / (float)UEN;
    float var = s2[0] / (float)UEN - mean * mean;   // biased (training-mode BN)
    float sc = gamma[f] * rsqrtf(var + BN_EPS);
    scale[f] = sc;
    shift[f] = beta[f] - mean * sc;
  }
}

// wscaled[a,f] = w[a,f]*scale[f] (f16, zero pad a>=105); constA[a] = sum_f w[a,f]*shift[f]
__global__ void k_ua_prep(const float* __restrict__ w, const float* __restrict__ scale,
                          const float* __restrict__ shift,
                          _Float16* __restrict__ wsc, float* __restrict__ constA) {
  __shared__ float sh[256];
  int a = blockIdx.x, f = threadIdx.x;
  float wv = (a < APN) ? w[(size_t)a * F2 + f] : 0.0f;
  wsc[(size_t)a * F2 + f] = (_Float16)(wv * scale[f]);
  sh[f] = wv * shift[f];
  __syncthreads();
  for (int off = 128; off > 0; off >>= 1) {
    if (f < off) sh[f] += sh[f + off];
    __syncthreads();
  }
  if (f == 0) constA[a] = sh[0];
}

// PC head: hidden = W_ap1[256,256] @ Xap2[256,105] + bais_ap; pc = sigmoid(w_ap2 @ hidden)
__global__ void k_pc_head(const float* __restrict__ x5, const float* __restrict__ x6,
                          const float* __restrict__ w1, const float* __restrict__ w2,
                          const float* __restrict__ bap, float* __restrict__ outPC) {
  __shared__ float sh[256];
  int r = threadIdx.x;  // 256 threads
  for (int i = 0; i < APN; ++i) {
    float h = 0.0f;
    for (int f = 0; f < F2; ++f) {
      float xv = (f < HH) ? x5[(size_t)f * APNP + i] : x6[(size_t)(f - HH) * APNP + i];
      h += w1[(size_t)r * F2 + f] * xv;
    }
    h += bap[i];
    sh[r] = w2[r] * h;
    __syncthreads();
    for (int off = 128; off > 0; off >>= 1) {
      if (r < off) sh[r] += sh[r + off];
      __syncthreads();
    }
    if (r == 0) outPC[i] = 1.0f / (1.0f + __expf(-sh[0]));
    __syncthreads();
  }
}

// sum over rows of unbiased variance of weight_ue rows
__global__ void k_var_ue(const float* __restrict__ w, float* __restrict__ out) {
  __shared__ float sh[128];
  int t = threadIdx.x;  // 128 threads
  float acc = 0.0f;
  for (int a = t; a < APN; a += 128) {
    float s = 0.0f, s2 = 0.0f;
    for (int f = 0; f < F2; ++f) { float v = w[(size_t)a * F2 + f]; s += v; s2 += v * v; }
    float mean = s / (float)F2;
    acc += (s2 - (float)F2 * mean * mean) / (float)(F2 - 1);
  }
  sh[t] = acc;
  __syncthreads();
  for (int off = 64; off > 0; off >>= 1) {
    if (t < off) sh[t] += sh[t + off];
    __syncthreads();
  }
  if (t == 0) out[0] = sh[0];
}

// Per-UE row: UA_ori (transpose), min/max normalize, two softmaxes. One wave/row.
__global__ void k_finalize(const float* __restrict__ UA, float* __restrict__ out) {
  const int lane = threadIdx.x & 31;
  const int wave = threadIdx.x >> 5;
  const int u = blockIdx.x * 8 + wave;
  if (u >= UEN) return;
  float v[4];
  float mx = -3.4e38f, mn = 3.4e38f;
#pragma unroll
  for (int t = 0; t < 4; ++t) {
    int a = lane + 32 * t;
    float x = 0.0f;
    if (a < APN) {
      x = UA[(size_t)a * UEN + u];
      mx = fmaxf(mx, x);
      mn = fminf(mn, x);
    }
    v[t] = x;
  }
  for (int off = 16; off > 0; off >>= 1) {
    mx = fmaxf(mx, __shfl_xor(mx, off, 32));
    mn = fminf(mn, __shfl_xor(mn, off, 32));
  }
  float inv = 1.0f / (mx - mn);
  float ed[4], es[4], sd = 0.0f, ss = 0.0f;
#pragma unroll
  for (int t = 0; t < 4; ++t) {
    int a = lane + 32 * t;
    float nor = (v[t] - mx) * inv;               // in [-1, 0]; max term exp(0)=1
    float d = (a < APN) ? __expf(nor * 1000.0f) : 0.0f;
    float s = (a < APN) ? __expf(nor) : 0.0f;
    ed[t] = d; es[t] = s; sd += d; ss += s;
  }
  for (int off = 16; off > 0; off >>= 1) {
    sd += __shfl_xor(sd, off, 32);
    ss += __shfl_xor(ss, off, 32);
  }
  float rd = 1.0f / sd, rs = 1.0f / ss;
#pragma unroll
  for (int t = 0; t < 4; ++t) {
    int a = lane + 32 * t;
    if (a < APN) {
      out[(size_t)u * APN + a] = v[t];
      out[(size_t)(UEN * APN) + (size_t)u * APN + a] = ed[t] * rd;
      out[(size_t)(2 * UEN * APN) + (size_t)u * APN + a] = es[t] * rs;
    }
  }
}

// ---------------------------------------------------------------------------
extern "C" void kernel_launch(void* const* d_in, const int* in_sizes, int n_in,
                              void* d_out, int out_size, void* d_ws, size_t ws_size,
                              hipStream_t stream) {
  (void)in_sizes; (void)n_in; (void)out_size; (void)ws_size;

  const float* pl    = (const float*)d_in[0];
  const int*   adjue = (const int*)d_in[2];     // [UEN, S], values < APN
  const int*   adjap = (const int*)d_in[3];     // [APN, S], values < UEN
  const float* W1ap1 = (const float*)d_in[4];   // [128, 4105]
  const float* W1ap2 = (const float*)d_in[5];   // [128, 8000]
  const float* W1ue1 = (const float*)d_in[6];   // [128, 4105]
  const float* W1ue2 = (const float*)d_in[7];   // [128, 210]
  const float* W2ap1 = (const float*)d_in[8];   // [128, 512]
  const float* W2ap2 = (const float*)d_in[9];
  const float* W2ue1 = (const float*)d_in[10];
  const float* W2ue2 = (const float*)d_in[11];
  const float* wap1  = (const float*)d_in[12];  // [256, 256]
  const float* wap2  = (const float*)d_in[13];  // [1, 256]
  const float* bap   = (const float*)d_in[14];  // [105]
  const float* wue   = (const float*)d_in[15];  // [105, 256]
  const float* bue   = (const float*)d_in[16];  // [4000]
  const float* gam   = (const float*)d_in[17];
  const float* bet   = (const float*)d_in[18];
  float* out = (float*)d_out;

  // ---- workspace carve-up (256B aligned) ----
  char* base = (char*)d_ws;
  size_t off = 0;
  auto alloc = [&](size_t bytes) -> void* {
    off = (off + 255) & ~(size_t)255;
    void* p = base + off;
    off += bytes;
    return p;
  };
  int* ap2ap = (int*)alloc(sizeof(int) * APN * SS);
  int* ue2ue = (int*)alloc(sizeof(int) * UEN * SS);
  _Float16* pl_h  = (_Float16*)alloc(2ull * APNP * UEN);  // [112, 4000]
  _Float16* plT_h = (_Float16*)alloc(2ull * UEN * K105);  // [4000, 128]
  _Float16* A1h = (_Float16*)alloc(2ull * HH * UEN);
  _Float16* B1h = (_Float16*)alloc(2ull * HH * K105);
  _Float16* A2h = (_Float16*)alloc(2ull * HH * UEN);
  _Float16* B2h = (_Float16*)alloc(2ull * HH * UEN);
  _Float16* A3h = (_Float16*)alloc(2ull * HH * K105);
  _Float16* B3h = (_Float16*)alloc(2ull * HH * UEN);
  _Float16* A4h = (_Float16*)alloc(2ull * HH * K105);
  _Float16* B4h = (_Float16*)alloc(2ull * HH * K105);
  _Float16* L2h[8];
  for (int i = 0; i < 8; ++i) L2h[i] = (_Float16*)alloc(2ull * HH * F2);
  _Float16* Xap_h  = (_Float16*)alloc(2ull * APNP * F2);
  _Float16* Xue_h  = (_Float16*)alloc(2ull * UEN * F2);
  _Float16* Xue2_h = (_Float16*)alloc(2ull * UEN * F2);
  _Float16* wsc_h  = (_Float16*)alloc(2ull * APNP * F2);
  auto fbuf = [&](size_t n) { return (float*)alloc(4ull * n); };
  float* G1 = fbuf(HH * APNP); float* P1 = fbuf(HH * UEN);
  float* G2 = fbuf(HH * APNP); float* P2 = fbuf(HH * APNP);
  float* G3 = fbuf(HH * UEN);  float* P3 = fbuf(HH * APNP);
  float* G4 = fbuf(HH * UEN);  float* P4 = fbuf(HH * UEN);
  float* x1 = fbuf(HH * APNP); float* x2 = fbuf(HH * APNP);
  float* x3 = fbuf(HH * UEN);  float* x4 = fbuf(HH * UEN);
  float* G5 = fbuf(HH * APNP); float* P5 = fbuf(HH * UEN);
  float* G6 = fbuf(HH * APNP); float* P6 = fbuf(HH * APNP);
  float* G7 = fbuf(HH * UEN);  float* P7 = fbuf(HH * APNP);
  float* G8 = fbuf(HH * UEN);  float* P8 = fbuf(HH * UEN);
  float* x5 = fbuf(HH * APNP); float* x6 = fbuf(HH * APNP);
  float* x7 = fbuf(HH * UEN);  float* x8 = fbuf(HH * UEN);
  float* scale = fbuf(F2); float* shift = fbuf(F2); float* constA = fbuf(APNP);
  float* UA = fbuf(APNP * UEN);

  auto grid1 = [](int n, int b) { return (n + b - 1) / b; };

  // ---- two-hop adjacency ----
  k_two_hop<<<grid1(APN * SS, 256), 256, 0, stream>>>(adjap, adjue, APN * SS, ap2ap);
  k_two_hop<<<grid1(UEN * SS, 256), 256, 0, stream>>>(adjue, adjap, UEN * SS, ue2ue);

  // ---- f16 staging (zero-padded) ----
  auto stage = [&](const float* src, int ld, int c0, _Float16* dst, int rows,
                   int cols, int dR, int dC) {
    k_stage_f16<<<grid1(dR * dC, 256), 256, 0, stream>>>(src, ld, c0, dst, rows, cols, dR, dC);
  };
  stage(pl, UEN, 0, pl_h, APN, UEN, APNP, UEN);
  k_stageT_f16<<<grid1(UEN * K105, 256), 256, 0, stream>>>(pl, APN, UEN, plT_h, UEN, K105);
  stage(W1ap1, 4105, 0,    A1h, HH, UEN, HH, UEN);
  stage(W1ap1, 4105, 4000, B1h, HH, APN, HH, K105);
  stage(W1ap2, 8000, 0,    A2h, HH, UEN, HH, UEN);
  stage(W1ap2, 8000, 4000, B2h, HH, UEN, HH, UEN);
  stage(W1ue1, 4105, 0,    A3h, HH, APN, HH, K105);
  stage(W1ue1, 4105, 105,  B3h, HH, UEN, HH, UEN);
  stage(W1ue2, 210, 0,     A4h, HH, APN, HH, K105);
  stage(W1ue2, 210, 105,   B4h, HH, APN, HH, K105);
  const float* W2s[4] = {W2ap1, W2ap2, W2ue1, W2ue2};
  for (int i = 0; i < 4; ++i) {
    stage(W2s[i], 512, 0,   L2h[2 * i],     HH, F2, HH, F2);  // self half
    stage(W2s[i], 512, 256, L2h[2 * i + 1], HH, F2, HH, F2);  // cross half
  }

  // ---- layer-1 WMMA GEMMs ----
  dim3 blkM128(32 * (HH / 16));   // 256 threads, 8 waves
  dim3 blkM112(32 * (APNP / 16)); // 224 threads, 7 waves
  dim3 gN112(APNP / 16), gN4000(UEN / 16);
  k_gemm_abt_wmma<<<gN112,  blkM128, 0, stream>>>(A1h, pl_h,  G1, UEN,  APNP, nullptr, nullptr);
  k_gemm_abt_wmma<<<gN4000, blkM128, 0, stream>>>(B1h, plT_h, P1, K105, UEN,  nullptr, nullptr);
  k_gemm_abt_wmma<<<gN112,  blkM128, 0, stream>>>(A2h, pl_h,  G2, UEN,  APNP, nullptr, nullptr);
  k_gemm_abt_wmma<<<gN112,  blkM128, 0, stream>>>(B2h, pl_h,  P2, UEN,  APNP, nullptr, nullptr);
  k_gemm_abt_wmma<<<gN4000, blkM128, 0, stream>>>(A3h, plT_h, G3, K105, UEN,  nullptr, nullptr);
  k_gemm_abt_wmma<<<gN112,  blkM128, 0, stream>>>(B3h, pl_h,  P3, UEN,  APNP, nullptr, nullptr);
  k_gemm_abt_wmma<<<gN4000, blkM128, 0, stream>>>(A4h, plT_h, G4, K105, UEN,  nullptr, nullptr);
  k_gemm_abt_wmma<<<gN4000, blkM128, 0, stream>>>(B4h, plT_h, P4, K105, UEN,  nullptr, nullptr);

  // ---- gather + relu epilogues ----
  auto combine = [&](const float* G, const float* P, int ldP, const int* idx,
                     float* o, int ldN, int Nv) {
    k_combine_gather<<<grid1(HH * ldN, 256), 256, 0, stream>>>(G, P, ldP, idx, o, ldN, Nv);
  };
  combine(G1, P1, UEN,  adjap, x1, APNP, APN);
  combine(G2, P2, APNP, ap2ap, x2, APNP, APN);
  combine(G3, P3, APNP, adjue, x3, UEN,  UEN);
  combine(G4, P4, UEN,  ue2ue, x4, UEN,  UEN);

  // ---- layer-2 feature staging + GEMMs ----
  k_stage_concatT<<<grid1(APNP * F2, 256), 256, 0, stream>>>(x1, x2, APNP, Xap_h, APNP);
  k_stage_concatT<<<grid1(UEN * F2, 256), 256, 0, stream>>>(x3, x4, UEN, Xue_h, UEN);
  k_gemm_abt_wmma<<<gN112,  blkM128, 0, stream>>>(L2h[0], Xap_h, G5, F2, APNP, nullptr, nullptr);
  k_gemm_abt_wmma<<<gN4000, blkM128, 0, stream>>>(L2h[1], Xue_h, P5, F2, UEN,  nullptr, nullptr);
  k_gemm_abt_wmma<<<gN112,  blkM128, 0, stream>>>(L2h[2], Xap_h, G6, F2, APNP, nullptr, nullptr);
  k_gemm_abt_wmma<<<gN112,  blkM128, 0, stream>>>(L2h[3], Xap_h, P6, F2, APNP, nullptr, nullptr);
  k_gemm_abt_wmma<<<gN4000, blkM128, 0, stream>>>(L2h[4], Xue_h, G7, F2, UEN,  nullptr, nullptr);
  k_gemm_abt_wmma<<<gN112,  blkM128, 0, stream>>>(L2h[5], Xap_h, P7, F2, APNP, nullptr, nullptr);
  k_gemm_abt_wmma<<<gN4000, blkM128, 0, stream>>>(L2h[6], Xue_h, G8, F2, UEN,  nullptr, nullptr);
  k_gemm_abt_wmma<<<gN4000, blkM128, 0, stream>>>(L2h[7], Xue_h, P8, F2, UEN,  nullptr, nullptr);
  combine(G5, P5, UEN,  adjap, x5, APNP, APN);
  combine(G6, P6, APNP, ap2ap, x6, APNP, APN);
  combine(G7, P7, APNP, adjue, x7, UEN,  UEN);
  combine(G8, P8, UEN,  ue2ue, x8, UEN,  UEN);

  // ---- PC head -> d_out[1260000 .. 1260104] ----
  k_pc_head<<<1, 256, 0, stream>>>(x5, x6, wap1, wap2, bap, out + 3 * UEN * APN);

  // ---- BN folded into UA GEMM ----
  k_bn_stats<<<F2, 256, 0, stream>>>(x7, x8, gam, bet, scale, shift);
  k_stage_concatT<<<grid1(UEN * F2, 256), 256, 0, stream>>>(x7, x8, UEN, Xue2_h, UEN);
  k_ua_prep<<<APNP, 256, 0, stream>>>(wue, scale, shift, wsc_h, constA);
  k_gemm_abt_wmma<<<gN4000, blkM112, 0, stream>>>(wsc_h, Xue2_h, UA, F2, UEN, constA, bue);

  // ---- normalize + softmaxes -> d_out[0 .. 1259999] ----
  k_finalize<<<grid1(UEN, 8), 256, 0, stream>>>(UA, out);

  // ---- sum of unbiased row variances of weight_ue -> d_out[1260105] ----
  k_var_ue<<<1, 128, 0, stream>>>(wue, out + 3 * UEN * APN + APN);
}
